// DGC_36644660969475
// MI455X (gfx1250) — matrix operand order repaired and checked by reading, not compiled, not run on hardware
//
#include <hip/hip_runtime.h>

// ---------------------------------------------------------------------------
// CDNA5 / gfx1250 SDCN-style forward pass.
// All heavy math -> v_wmma_f32_16x16x32_bf16 (f32 accumulate).
// adj (10000x10000) converted once to bf16 padded to K=10016 (200MB, ~fits
// the 192MB L2); the five adj GEMMs (~0.63 of ~0.7 total TFLOP) then run
// compute-bound out of L2. K padded to a multiple of 32 everywhere so the
// WMMA K-loop has zero bounds checks: 2x global_load_b128 per fragment.
// K-loop is register double-buffered: fragments for step kk+32 are issued
// before the WMMAs of step kk, so the compiler waits on loadcnt > 0 instead
// of draining the load pipe before every matrix op.
// Each wave computes a 16x64 output strip (4 WMMAs per A fragment).
// AE decoder is dead code w.r.t. the returned tuple and is skipped.
// ---------------------------------------------------------------------------

typedef __bf16 bf16_t;
typedef __attribute__((ext_vector_type(16))) __bf16 v16bf;
typedef __attribute__((ext_vector_type(8)))  float  v8f;

// ------------------------- elementwise kernels -----------------------------

__global__ void k_f32_to_bf16(const float* __restrict__ in, bf16_t* __restrict__ out,
                              size_t n, int relu) {
  size_t i = (size_t)blockIdx.x * blockDim.x + threadIdx.x;
  if (i >= n) return;
  float v = in[i];
  if (relu) v = v > 0.f ? v : 0.f;
  out[i] = (bf16_t)v;
}

// convert [rows x cols] f32 -> [rows x Kp] bf16, zero-filling cols..Kp-1
__global__ void k_f32_to_bf16_pad(const float* __restrict__ in, bf16_t* __restrict__ out,
                                  int rows, int cols, int Kp) {
  size_t i = (size_t)blockIdx.x * blockDim.x + threadIdx.x;
  size_t total = (size_t)rows * Kp;
  if (i >= total) return;
  int r = (int)(i / Kp);
  int c = (int)(i % Kp);
  out[i] = (c < cols) ? (bf16_t)in[(size_t)r * cols + c] : (bf16_t)0.f;
}

// zero pad columns Kvalid..Kp-1 of a [rows x Kp] bf16 buffer
__global__ void k_zero_pad(bf16_t* __restrict__ buf, int rows, int Kp, int Kvalid) {
  int pad = Kp - Kvalid;
  int total = rows * pad;
  int i = blockIdx.x * blockDim.x + threadIdx.x;
  if (i >= total) return;
  int r = i / pad, c = i % pad;
  buf[(size_t)r * Kp + Kvalid + c] = (bf16_t)0.f;
}

// in: K x N (row major), out: N x K (row major)  -- weight convert+transpose
__global__ void k_f32_to_bf16_tr(const float* __restrict__ in, bf16_t* __restrict__ out,
                                 int K, int N) {
  size_t i = (size_t)blockIdx.x * blockDim.x + threadIdx.x;
  size_t total = (size_t)K * (size_t)N;
  if (i >= total) return;
  int nn = (int)(i / (size_t)K);
  int kk = (int)(i % (size_t)K);
  out[i] = (bf16_t)in[(size_t)kk * N + nn];
}

// out = bf16( sa * (relu_a ? max(a,0) : a) + sb * b )
__global__ void k_mix_to_bf16(const float* __restrict__ a, const float* __restrict__ b,
                              bf16_t* __restrict__ out, size_t n,
                              float sa, float sb, int relu_a) {
  size_t i = (size_t)blockIdx.x * blockDim.x + threadIdx.x;
  if (i >= n) return;
  float av = a[i];
  if (relu_a) av = av > 0.f ? av : 0.f;
  out[i] = (bf16_t)(sa * av + sb * b[i]);
}

// row-wise softmax over 16 columns
__global__ void k_softmax16(const float* __restrict__ in, float* __restrict__ out, int n) {
  int i = blockIdx.x * blockDim.x + threadIdx.x;
  if (i >= n) return;
  const float* p = in + (size_t)i * 16;
  float m = p[0];
#pragma unroll
  for (int j = 1; j < 16; ++j) m = fmaxf(m, p[j]);
  float e[16], s = 0.f;
#pragma unroll
  for (int j = 0; j < 16; ++j) { e[j] = __expf(p[j] - m); s += e[j]; }
  float inv = 1.f / s;
#pragma unroll
  for (int j = 0; j < 16; ++j) out[(size_t)i * 16 + j] = e[j] * inv;
}

// Student-t soft assignment; V=1 => exponent (V+1)/2 == 1, q = (1+d2)^-1 norm.
__global__ void k_cluster_q(const float* __restrict__ h, const float* __restrict__ cluster,
                            float* __restrict__ q, int n) {
  int i = blockIdx.x * blockDim.x + threadIdx.x;
  if (i >= n) return;
  const float* hp = h + (size_t)i * 64;
  float hv[64];
#pragma unroll
  for (int j = 0; j < 64; ++j) hv[j] = hp[j];
  float qv[16], s = 0.f;
  for (int c = 0; c < 16; ++c) {
    const float* cp = cluster + c * 64;
    float d2 = 0.f;
#pragma unroll
    for (int j = 0; j < 64; ++j) { float t = hv[j] - cp[j]; d2 += t * t; }
    float v = 1.f / (1.f + d2);
    qv[c] = v; s += v;
  }
  float inv = 1.f / s;
#pragma unroll
  for (int c = 0; c < 16; ++c) q[(size_t)i * 16 + c] = qv[c] * inv;
}

// ------------------------------ WMMA GEMM ----------------------------------
// C[M,N] = A[M,K] (bf16 row major, K multiple of 32, no tail)
//          x Bt[N,K] (bf16, B stored transposed)
// One wave computes a 16 x (16*NT) strip: one A fragment, NT WMMAs per K-step.
// flags: bit0 = relu epilogue, bit1 = bf16 output stored transposed [N,ldcb].
union frag32 { uint4 q[2]; v16bf v; };

__device__ __forceinline__ void load_fragA(frag32& f, const bf16_t* p) {
  // A 16x32 bf16 fragment (ISA 7.12.2): half 0 -> K {0..7,16..23},
  // half 1 -> K {8..15,24..31}; two aligned 16B runs -> global_load_b128.
  f.q[0] = *(const uint4*)(p);
  f.q[1] = *(const uint4*)(p + 16);
}
__device__ __forceinline__ void load_fragB(frag32& f, const bf16_t* p) {
  // B 32x16 fragment: lane = column; half 0 -> K {0..7,16..23},
  // half 1 -> K {8..15,24..31} (matches A), since B is stored transposed [N,K].
  f.q[0] = *(const uint4*)(p);
  f.q[1] = *(const uint4*)(p + 16);
}

template <int NT>
__global__ __launch_bounds__(256) void k_gemm_bf16_wmma(
    const bf16_t* __restrict__ A, const bf16_t* __restrict__ Bt,
    const float* __restrict__ bias,
    float* __restrict__ Cf, bf16_t* __restrict__ Cb,
    int M, int N, int K, int ldcb, int flags) {
  const int lane = threadIdx.x & 31;
  const int wave = threadIdx.x >> 5;
  const int tiles_m = M >> 4;
  const int tm = blockIdx.y * 8 + wave;
  if (tm >= tiles_m) return;          // wave-uniform: EXEC stays all-ones
  const int m0 = tm << 4;
  const int n0 = blockIdx.x * (16 * NT);
  const int half = lane >> 4;         // 0: lanes 0-15, 1: lanes 16-31
  const int lid  = lane & 15;

  const bf16_t* Ap = A + (size_t)(m0 + lid) * K + half * 8;   // A frag base
  const bf16_t* Bp[NT];
#pragma unroll
  for (int t = 0; t < NT; ++t)
    Bp[t] = Bt + (size_t)(n0 + 16 * t + lid) * K + half * 8;  // B frag base

  v8f acc[NT];
#pragma unroll
  for (int t = 0; t < NT; ++t) acc[t] = (v8f){0.f,0.f,0.f,0.f,0.f,0.f,0.f,0.f};

  // -------- software-pipelined K loop (register double buffering) ----------
  frag32 a_cur, b_cur[NT];
  load_fragA(a_cur, Ap);
#pragma unroll
  for (int t = 0; t < NT; ++t) load_fragB(b_cur[t], Bp[t]);

  int kk = 0;
  while (kk + 32 < K) {
    const int kn = kk + 32;
    __builtin_prefetch(Ap + kn + 512, 0, 0);        // -> global_prefetch_b8
    __builtin_prefetch(Bp[0] + kn + 512, 0, 0);
    // issue next-step loads first: WMMAs below depend only on older loads,
    // so the scheduler can overlap these loads with the matrix pipe.
    frag32 a_nxt, b_nxt[NT];
    load_fragA(a_nxt, Ap + kn);
#pragma unroll
    for (int t = 0; t < NT; ++t) load_fragB(b_nxt[t], Bp[t] + kn);
    // compute current step
#pragma unroll
    for (int t = 0; t < NT; ++t)
      acc[t] = __builtin_amdgcn_wmma_f32_16x16x32_bf16(
          false, a_cur.v, false, b_cur[t].v, (short)0, acc[t], false, false);
    a_cur = a_nxt;
#pragma unroll
    for (int t = 0; t < NT; ++t) b_cur[t] = b_nxt[t];
    kk = kn;
  }
#pragma unroll
  for (int t = 0; t < NT; ++t)
    acc[t] = __builtin_amdgcn_wmma_f32_16x16x32_bf16(
        false, a_cur.v, false, b_cur[t].v, (short)0, acc[t], false, false);

  // -------------------------------- epilogue -------------------------------
  const int mbase = m0 + half * 8;    // C/D layout: VGPR r -> M = r + 8*half
#pragma unroll
  for (int t = 0; t < NT; ++t) {
    const int ncol = n0 + 16 * t + lid;
    const float bv = bias ? bias[ncol] : 0.f;
#pragma unroll
    for (int r = 0; r < 8; ++r) {
      float v = acc[t][r] + bv;
      if (flags & 1) v = v > 0.f ? v : 0.f;
      const int m = mbase + r;
      if (Cf) Cf[(size_t)m * N + ncol] = v;
      if (Cb) {
        bf16_t x = (bf16_t)v;
        if (flags & 2) Cb[(size_t)ncol * ldcb + m] = x;  // transposed: next B
        else           Cb[(size_t)m * N + ncol] = x;     // row major:  next A
      }
    }
  }
}

// ------------------------------- launcher ----------------------------------

static inline void gemm(const bf16_t* A, const bf16_t* Bt, const float* bias,
                        float* Cf, bf16_t* Cb, int M, int N, int K, int ldcb,
                        int flags, hipStream_t s) {
  dim3 grid(1, ((M / 16) + 7) / 8);
  if (N % 64 == 0) {
    grid.x = N / 64;
    k_gemm_bf16_wmma<4><<<grid, 256, 0, s>>>(A, Bt, bias, Cf, Cb, M, N, K, ldcb, flags);
  } else {
    grid.x = N / 16;
    k_gemm_bf16_wmma<1><<<grid, 256, 0, s>>>(A, Bt, bias, Cf, Cb, M, N, K, ldcb, flags);
  }
}

static inline void cvt(const float* in, bf16_t* out, size_t n, int relu, hipStream_t s) {
  k_f32_to_bf16<<<(unsigned)((n + 255) / 256), 256, 0, s>>>(in, out, n, relu);
}
static inline void cvtT(const float* in, bf16_t* out, int K, int N, hipStream_t s) {
  size_t n = (size_t)K * N;
  k_f32_to_bf16_tr<<<(unsigned)((n + 255) / 256), 256, 0, s>>>(in, out, K, N);
}
static inline void mix(const float* a, const float* b, bf16_t* out, size_t n,
                       float sa, float sb, int relu_a, hipStream_t s) {
  k_mix_to_bf16<<<(unsigned)((n + 255) / 256), 256, 0, s>>>(a, b, out, n, sa, sb, relu_a);
}

extern "C" void kernel_launch(void* const* d_in, const int* in_sizes, int n_in,
                              void* d_out, int out_size, void* d_ws, size_t ws_size,
                              hipStream_t stream) {
  (void)in_sizes; (void)n_in; (void)out_size; (void)ws_size;
  const int   Nn  = 10000;
  const int   Mp  = 10016;            // Nn padded to a multiple of 32 (K dim)
  const int   NIN = 1024, E1 = 512, E2 = 512, E3 = 2048, NZ = 64, KC = 16;
  const float SG  = 0.3f, SA = 1.f - SG;

  // inputs (setup_inputs dict order); AE decoder params (10..17) are dead code
  const float* x       = (const float*)d_in[0];
  const float* adj     = (const float*)d_in[1];
  const float* enc1_w  = (const float*)d_in[2];
  const float* enc1_b  = (const float*)d_in[3];
  const float* enc2_w  = (const float*)d_in[4];
  const float* enc2_b  = (const float*)d_in[5];
  const float* enc3_w  = (const float*)d_in[6];
  const float* enc3_b  = (const float*)d_in[7];
  const float* zw      = (const float*)d_in[8];
  const float* zb      = (const float*)d_in[9];
  const float* g1w     = (const float*)d_in[18];
  const float* g2w     = (const float*)d_in[19];
  const float* g3w     = (const float*)d_in[20];
  const float* g4w     = (const float*)d_in[21];
  const float* g5w     = (const float*)d_in[22];
  const float* fc1w    = (const float*)d_in[23];
  const float* fc1b    = (const float*)d_in[24];
  const float* cluster = (const float*)d_in[25];

  // output layout: x_bar | q | predict | z
  float* out_xbar = (float*)d_out;
  float* out_q    = out_xbar + (size_t)Nn * NIN;
  float* out_pred = out_q    + (size_t)Nn * KC;
  float* out_z    = out_pred + (size_t)Nn * KC;

  // linear carve of d_ws (deterministic every call)
  char* wp = (char*)d_ws;
  auto carve = [&](size_t bytes) -> void* {
    void* r = (void*)wp;
    wp += (bytes + 255) & ~(size_t)255;
    return r;
  };
  bf16_t* adjb   = (bf16_t*)carve((size_t)Nn * Mp * 2);   // [Nn x Mp] padded
  bf16_t* xb     = (bf16_t*)carve((size_t)Nn * NIN * 2);
  bf16_t* enc1t  = (bf16_t*)carve((size_t)NIN * E1 * 2);
  bf16_t* enc2t  = (bf16_t*)carve((size_t)E1 * E2 * 2);
  bf16_t* enc3t  = (bf16_t*)carve((size_t)E2 * E3 * 2);
  bf16_t* zwt    = (bf16_t*)carve((size_t)E3 * NZ * 2);
  bf16_t* g1t    = (bf16_t*)carve((size_t)NIN * E1 * 2);
  bf16_t* g2t    = (bf16_t*)carve((size_t)E1 * E2 * 2);
  bf16_t* g3t    = (bf16_t*)carve((size_t)E2 * E3 * 2);
  bf16_t* g4t    = (bf16_t*)carve((size_t)E3 * NZ * 2);
  bf16_t* g5t    = (bf16_t*)carve((size_t)NZ * KC * 2);
  bf16_t* fc1t   = (bf16_t*)carve((size_t)NZ * NIN * 2);
  float*  tra1f  = (float*) carve((size_t)Nn * E1 * 4);
  bf16_t* tra1b  = (bf16_t*)carve((size_t)Nn * E1 * 2);
  float*  tra2f  = (float*) carve((size_t)Nn * E2 * 4);
  bf16_t* tra2b  = (bf16_t*)carve((size_t)Nn * E2 * 2);
  float*  tra3f  = (float*) carve((size_t)Nn * E3 * 4);
  bf16_t* tra3b  = (bf16_t*)carve((size_t)Nn * E3 * 2);
  bf16_t* tT512  = (bf16_t*)carve((size_t)E1 * Mp * 2);   // t1^T / t2^T (alias)
  float*  h512   = (float*) carve((size_t)Nn * E1 * 4);   // h1 / h2     (alias)
  bf16_t* m512   = (bf16_t*)carve((size_t)Nn * E1 * 2);   // m1 / m2     (alias)
  bf16_t* t3b    = (bf16_t*)carve((size_t)E3 * Mp * 2);
  float*  h3f    = (float*) carve((size_t)Nn * E3 * 4);
  bf16_t* m3b    = (bf16_t*)carve((size_t)Nn * E3 * 2);
  bf16_t* t4b    = (bf16_t*)carve((size_t)NZ * Mp * 2);
  float*  hf     = (float*) carve((size_t)Nn * NZ * 4);
  bf16_t* rhb    = (bf16_t*)carve((size_t)Nn * NZ * 2);
  bf16_t* m4b    = (bf16_t*)carve((size_t)Nn * NZ * 2);
  bf16_t* t5b    = (bf16_t*)carve((size_t)KC * Mp * 2);
  float*  h5f    = (float*) carve((size_t)Nn * KC * 4);

  // ---- conversions; adj gets zero-padded K columns (200MB bf16, ~L2-res) ----
  {
    size_t n = (size_t)Nn * Mp;
    k_f32_to_bf16_pad<<<(unsigned)((n + 255) / 256), 256, 0, stream>>>(adj, adjb, Nn, Nn, Mp);
  }
  cvt(x, xb, (size_t)Nn * NIN, 0, stream);
  cvtT(enc1_w, enc1t, NIN, E1, stream);
  cvtT(enc2_w, enc2t, E1,  E2, stream);
  cvtT(enc3_w, enc3t, E2,  E3, stream);
  cvtT(zw,     zwt,   E3,  NZ, stream);
  cvtT(g1w,    g1t,   NIN, E1, stream);
  cvtT(g2w,    g2t,   E1,  E2, stream);
  cvtT(g3w,    g3t,   E2,  E3, stream);
  cvtT(g4w,    g4t,   E3,  NZ, stream);
  cvtT(g5w,    g5t,   NZ,  KC, stream);
  cvtT(fc1w,   fc1t,  NZ,  NIN, stream);
  // zero the pad rows (m = Nn..Mp-1) of every transposed activation buffer
  k_zero_pad<<<(E1 * (Mp - Nn) + 255) / 256, 256, 0, stream>>>(tT512, E1, Mp, Nn);
  k_zero_pad<<<(E3 * (Mp - Nn) + 255) / 256, 256, 0, stream>>>(t3b,  E3, Mp, Nn);
  k_zero_pad<<<(NZ * (Mp - Nn) + 255) / 256, 256, 0, stream>>>(t4b,  NZ, Mp, Nn);
  k_zero_pad<<<(KC * (Mp - Nn) + 255) / 256, 256, 0, stream>>>(t5b,  KC, Mp, Nn);

  // ---- AE encoder ----
  gemm(xb,    enc1t, enc1_b, tra1f, tra1b, Nn, E1, NIN, 0, 1, stream);
  gemm(tra1b, enc2t, enc2_b, tra2f, tra2b, Nn, E2, E1,  0, 1, stream);
  gemm(tra2b, enc3t, enc3_b, tra3f, tra3b, Nn, E3, E2,  0, 1, stream);
  gemm(tra3b, zwt,   zb,     out_z, nullptr, Nn, NZ, E3, 0, 0, stream);
  // (AE decoder: dh1..dh3 / _x_bar_ae feed no output -> skipped)

  // ---- GNN stack with residual AE mixing ----
  gemm(xb, g1t, nullptr, nullptr, tT512, Nn, E1, NIN, Mp, 2, stream);    // t1^T
  gemm(adjb, tT512, nullptr, h512, nullptr, Nn, E1, Mp, 0, 1, stream);   // h1
  mix(h512, tra1f, m512, (size_t)Nn * E1, SA, SG, 0, stream);            // m1
  gemm(m512, g2t, nullptr, nullptr, tT512, Nn, E2, E1, Mp, 2, stream);   // t2^T
  gemm(adjb, tT512, nullptr, h512, nullptr, Nn, E2, Mp, 0, 1, stream);   // h2
  mix(h512, tra2f, m512, (size_t)Nn * E2, SA, SG, 0, stream);            // m2
  gemm(m512, g3t, nullptr, nullptr, t3b, Nn, E3, E2, Mp, 2, stream);     // t3^T
  gemm(adjb, t3b, nullptr, h3f, nullptr, Nn, E3, Mp, 0, 1, stream);      // h3
  mix(h3f, tra3f, m3b, (size_t)Nn * E3, SA, SG, 0, stream);              // m3
  gemm(m3b, g4t, nullptr, nullptr, t4b, Nn, NZ, E3, Mp, 2, stream);      // t4^T
  gemm(adjb, t4b, nullptr, hf, nullptr, Nn, NZ, Mp, 0, 0, stream);       // h

  cvt(hf, rhb, (size_t)Nn * NZ, 1, stream);                              // relu(h)
  mix(hf, out_z, m4b, (size_t)Nn * NZ, SA, SG, 1, stream);               // m4
  gemm(m4b, g5t, nullptr, nullptr, t5b, Nn, KC, NZ, Mp, 2, stream);      // t5^T
  gemm(adjb, t5b, nullptr, h5f, nullptr, Nn, KC, Mp, 0, 0, stream);      // h5

  // ---- heads ----
  k_softmax16<<<(Nn + 255) / 256, 256, 0, stream>>>(h5f, out_pred, Nn);
  gemm(rhb, fc1t, fc1b, out_xbar, nullptr, Nn, NIN, NZ, 0, 1, stream);   // x_bar
  k_cluster_q<<<(Nn + 255) / 256, 256, 0, stream>>>(hf, cluster, out_q, Nn);
}
